// BaseQuantizer_35828617183399
// MI455X (gfx1250) — matrix operand order repaired and checked by reading, not compiled, not run on hardware
//
#include <hip/hip_runtime.h>
#include <hip/hip_bf16.h>

typedef __attribute__((ext_vector_type(16))) _Float16 v16h;
typedef __attribute__((ext_vector_type(8)))  _Float16 v8h;
typedef __attribute__((ext_vector_type(8)))  float    v8f;

#define BQ_B 16
#define BQ_C 128
#define BQ_H 32
#define BQ_V 8192
#define BQ_NT (BQ_V / 16)                     // 512 code tiles
#define BQ_NPIX (BQ_B * BQ_C * BQ_H * BQ_H)   // 2,097,152
#define BQ_TMAX (BQ_B * BQ_H * BQ_H)          // 16,384 max tokens per level
#define BQ_PAD_ROWS 192                       // embh tail padding (prefetch + dbl-buffer overrun)

// ---------------------------------------------------------------------------
// K0: f16 codebook copy + ||e||^2 table
// ---------------------------------------------------------------------------
__global__ void bq_prep_emb(const float* __restrict__ emb,
                            _Float16* __restrict__ embh,
                            float* __restrict__ esq) {
    int v = blockIdx.x * blockDim.x + threadIdx.x;
    if (v >= BQ_V) return;
    float s = 0.f;
    #pragma unroll 8
    for (int k = 0; k < BQ_C; ++k) {
        float e = emb[v * BQ_C + k];
        s += e * e;
        embh[v * BQ_C + k] = (_Float16)e;
    }
    esq[v] = s;
}

// ---------------------------------------------------------------------------
// K1: z_rest = z_enc (stop_gradient copy), z_dec = 0
// ---------------------------------------------------------------------------
__global__ void bq_init_z(const float* __restrict__ zenc,
                          float* __restrict__ zrest,
                          float* __restrict__ zdec) {
    int id = blockIdx.x * blockDim.x + threadIdx.x;
    if (id >= BQ_NPIX) return;
    zrest[id] = zenc[id];
    zdec[id]  = 0.f;
}

// ---------------------------------------------------------------------------
// K2: area-downsample residual to [T, C] f16 queries (NHWC token-major)
// ---------------------------------------------------------------------------
__global__ void bq_downsample(const float* __restrict__ zrest,
                              _Float16* __restrict__ qh, int pn) {
    int id = blockIdx.x * blockDim.x + threadIdx.x;
    int total = BQ_B * pn * pn * BQ_C;
    if (id >= total) return;
    int c = id & (BQ_C - 1);
    int t = id >> 7;
    int x = t % pn;
    int y = (t / pn) % pn;
    int b = t / (pn * pn);
    int f = BQ_H / pn;
    const float* base = zrest + ((b * BQ_C + c) * BQ_H + y * f) * BQ_H + x * f;
    float s = 0.f;
    for (int i = 0; i < f; ++i)
        for (int j = 0; j < f; ++j)
            s += base[i * BQ_H + j];
    qh[id] = (_Float16)(s / (float)(f * f));
}

// ---------------------------------------------------------------------------
// K3: WMMA nearest-code search.
//   dist' = ||e||^2 - 2 * (z . e)   (||z||^2 dropped: constant per row)
//   One wave owns 32 queries (two 16x16 C tiles, A fragments VGPR-resident),
//   sweeps 512 16-code tiles with K=128 as four 16x16x32 f16 WMMAs each.
//   B tiles are register double-buffered (unroll-by-2, no copy movs) and
//   prefetched 8 tiles ahead so WMMAs never stall on WGP$/L2 latency.
// ---------------------------------------------------------------------------
#define SHUF16(lo, hi) __builtin_shufflevector(lo, hi, 0,1,2,3,4,5,6,7,8,9,10,11,12,13,14,15)

__device__ __forceinline__ void bq_loadB(v16h Bt[4], const _Float16* p) {
    #pragma unroll
    for (int c = 0; c < 4; ++c)
        Bt[c] = *(const v16h*)(p + c * 32);
}

__device__ __forceinline__ void bq_tile(const v16h A0[4], const v16h A1[4],
                                        const v16h Bt[4], float es, int codev,
                                        float mv0[8], int mi0[8],
                                        float mv1[8], int mi1[8]) {
    v8f acc0 = {}; v8f acc1 = {};
    #pragma unroll
    for (int c = 0; c < 4; ++c) {
        acc0 = __builtin_amdgcn_wmma_f32_16x16x32_f16(false, A0[c], false, Bt[c],
                                                      (short)0, acc0, false, false);
        acc1 = __builtin_amdgcn_wmma_f32_16x16x32_f16(false, A1[c], false, Bt[c],
                                                      (short)0, acc1, false, false);
    }
    #pragma unroll
    for (int r = 0; r < 8; ++r) {
        float d0 = fmaf(-2.f, acc0[r], es);
        if (d0 < mv0[r]) { mv0[r] = d0; mi0[r] = codev; }
        float d1 = fmaf(-2.f, acc1[r], es);
        if (d1 < mv1[r]) { mv1[r] = d1; mi1[r] = codev; }
    }
}

__global__ __launch_bounds__(128)
void bq_argmin(const _Float16* __restrict__ qh,
               const _Float16* __restrict__ embh,
               const float* __restrict__ esq,
               int* __restrict__ tokens, int T) {
    const int lane = threadIdx.x & 31;
    const int wave = threadIdx.x >> 5;
    const int t0 = (blockIdx.x * 4 + wave) * 32;   // first query of tile pair
    if (t0 >= T) return;                           // uniform whole-wave exit
    const int t1 = t0 + 16;
    const int half = lane >> 4;                    // 0: rows 0-7 / K-lo, 1: rows 8-15 / K-hi
    const int l15 = lane & 15;

    // ---- A fragments (resident across the whole code sweep) ----
    // A chunks in this code's contiguous k order (B matches): lane<16 holds K 0..15, lane>=16 holds K 16..31
    int r0 = t0 + l15;
    int r1 = t1 + l15; if (r1 >= T) r1 = T - 1;    // clamp; results discarded
    const int ka = half ? 16 : 0;
    v16h A0[4], A1[4];
    #pragma unroll
    for (int c = 0; c < 4; ++c) {
        const _Float16* p0 = qh + r0 * BQ_C + c * 32 + ka;
        v8h lo0 = *(const v8h*)p0;
        v8h hi0 = *(const v8h*)(p0 + 8);
        A0[c] = SHUF16(lo0, hi0);
        const _Float16* p1 = qh + r1 * BQ_C + c * 32 + ka;
        v8h lo1 = *(const v8h*)p1;
        v8h hi1 = *(const v8h*)(p1 + 8);
        A1[c] = SHUF16(lo1, hi1);
    }

    float mv0[8], mv1[8];
    int   mi0[8], mi1[8];
    #pragma unroll
    for (int r = 0; r < 8; ++r) {
        mv0[r] = __builtin_inff(); mv1[r] = __builtin_inff();
        mi0[r] = 0;                mi1[r] = 0;
    }

    // ---- sweep code tiles, double-buffered in registers ----
    // B chunks, same contiguous k order as A: lane = column (code), lane<16 K 0..15, lane>=16 K 16..31
    const int kb = half ? 16 : 0;
    const size_t tstride = (size_t)16 * BQ_C;               // halves per code tile
    const _Float16* bp  = embh + (size_t)l15 * BQ_C + kb;   // this lane's tile-0 chunk base
    // prefetch pointer: half 0 warms bytes [0,128), half 1 bytes [128,256) of each row
    const _Float16* pfp = embh + (size_t)l15 * BQ_C + (half ? 64 : 0);
    const float* ep = esq + l15;

    v16h Ba[4], Bb[4];
    bq_loadB(Ba, bp);
    float ea = ep[0];

    for (int nt = 0; nt < BQ_NT; nt += 2) {
        // stage tile nt+1 while computing tile nt
        bq_loadB(Bb, bp + tstride);
        float eb = ep[16];
        __builtin_prefetch(pfp + (size_t)(nt + 8) * tstride, 0, 3);
        bq_tile(A0, A1, Ba, ea, nt * 16 + l15, mv0, mi0, mv1, mi1);

        // stage tile nt+2 (last iteration reads padded tail; result unused)
        bq_loadB(Ba, bp + 2 * tstride);
        ea = ep[32];
        __builtin_prefetch(pfp + (size_t)(nt + 9) * tstride, 0, 3);
        bq_tile(A0, A1, Bb, eb, (nt + 1) * 16 + l15, mv0, mi0, mv1, mi1);

        bp += 2 * tstride;
        ep += 32;
    }

    // ---- butterfly argmin across the 16 lanes of each half-wave ----
    #pragma unroll
    for (int r = 0; r < 8; ++r) {
        #pragma unroll
        for (int off = 8; off >= 1; off >>= 1) {
            float ov = __shfl_xor(mv0[r], off);
            int   oi = __shfl_xor(mi0[r], off);
            if (ov < mv0[r] || (ov == mv0[r] && oi < mi0[r])) { mv0[r] = ov; mi0[r] = oi; }
            ov = __shfl_xor(mv1[r], off);
            oi = __shfl_xor(mi1[r], off);
            if (ov < mv1[r] || (ov == mv1[r] && oi < mi1[r])) { mv1[r] = ov; mi1[r] = oi; }
        }
    }

    // C/D layout: VGPR r, lanes 0-15 -> row r; lanes 16-31 -> row r+8
    if (l15 == 0) {
        const int rb = half * 8;
        #pragma unroll
        for (int r = 0; r < 8; ++r) tokens[t0 + rb + r] = mi0[r];
        if (t1 < T) {
            #pragma unroll
            for (int r = 0; r < 8; ++r) tokens[t1 + rb + r] = mi1[r];
        }
    }
}

// ---------------------------------------------------------------------------
// K4: gather f32 code vectors for chosen tokens -> zq [B, pn, pn, C]
// ---------------------------------------------------------------------------
__global__ void bq_gather(const float* __restrict__ emb,
                          const int* __restrict__ tokens,
                          float* __restrict__ zq, int T) {
    int id = blockIdx.x * blockDim.x + threadIdx.x;
    if (id >= T * BQ_C) return;
    int t = id >> 7;
    int c = id & (BQ_C - 1);
    zq[id] = emb[tokens[t] * BQ_C + c];
}

// ---------------------------------------------------------------------------
// K5: bicubic upsample (PyTorch a=-0.75, border-clamped taps) fused with
//     z_dec += zu ; z_rest -= zu ; out[level] = z_dec
// ---------------------------------------------------------------------------
__device__ __forceinline__ float bq_cubw(float t) {
    const float a = -0.75f;
    t = fabsf(t);
    if (t <= 1.f) return ((a + 2.f) * t - (a + 3.f)) * t * t + 1.f;
    if (t < 2.f)  return ((((t - 5.f) * t + 8.f) * t) - 4.f) * a;
    return 0.f;
}

__global__ void bq_upsample_update(const float* __restrict__ zq,
                                   float* __restrict__ zdec,
                                   float* __restrict__ zrest,
                                   float* __restrict__ out, int pn) {
    int id = blockIdx.x * blockDim.x + threadIdx.x;
    if (id >= BQ_NPIX) return;
    int w = id & 31;
    int h = (id >> 5) & 31;
    int c = (id >> 10) & (BQ_C - 1);
    int b = id >> 17;

    float zu;
    if (pn == BQ_H) {
        // last level: identity (zq already at full resolution, NHWC)
        zu = zq[((b * BQ_H + h) * BQ_H + w) * BQ_C + c];
    } else {
        const float s = (float)pn / (float)BQ_H;
        float xh = ((float)h + 0.5f) * s - 0.5f;
        float xw = ((float)w + 0.5f) * s - 0.5f;
        int h0 = (int)floorf(xh);  float th = xh - (float)h0;
        int w0 = (int)floorf(xw);  float tw = xw - (float)w0;
        float wh[4], ww[4]; int ih[4], iw[4];
        #pragma unroll
        for (int o = 0; o < 4; ++o) {
            wh[o] = bq_cubw((float)(o - 1) - th);
            ww[o] = bq_cubw((float)(o - 1) - tw);
            int hi_ = h0 + o - 1; ih[o] = hi_ < 0 ? 0 : (hi_ > pn - 1 ? pn - 1 : hi_);
            int wi_ = w0 + o - 1; iw[o] = wi_ < 0 ? 0 : (wi_ > pn - 1 ? pn - 1 : wi_);
        }
        zu = 0.f;
        #pragma unroll
        for (int a_ = 0; a_ < 4; ++a_) {
            const float* row = zq + ((b * pn + ih[a_]) * pn) * BQ_C + c;
            float racc = 0.f;
            #pragma unroll
            for (int d = 0; d < 4; ++d)
                racc = fmaf(ww[d], row[iw[d] * BQ_C], racc);
            zu = fmaf(wh[a_], racc, zu);
        }
    }
    float nd = zdec[id] + zu;
    zdec[id]  = nd;
    zrest[id] -= zu;
    out[id]   = nd;
}

// ---------------------------------------------------------------------------
extern "C" void kernel_launch(void* const* d_in, const int* in_sizes, int n_in,
                              void* d_out, int out_size, void* d_ws, size_t ws_size,
                              hipStream_t stream) {
    const float* z_enc = (const float*)d_in[0];   // [16,128,32,32]
    const float* emb   = (const float*)d_in[1];   // [8192,128]
    float* out = (float*)d_out;                    // [6,16,128,32,32]

    char* ws = (char*)d_ws;
    size_t off = 0;
    auto carve = [&](size_t bytes) -> void* {
        void* p = ws + off;
        off += (bytes + 255) & ~(size_t)255;
        return p;
    };
    // embh/esq carry padded tails so the double-buffer overrun load and the
    // +8-tile prefetch always touch valid workspace memory.
    _Float16* embh   = (_Float16*)carve((size_t)(BQ_V + BQ_PAD_ROWS) * BQ_C * sizeof(_Float16));
    float*    esq    = (float*)   carve((size_t)(BQ_V + 64) * sizeof(float));
    float*    zrest  = (float*)   carve((size_t)BQ_NPIX * sizeof(float));
    float*    zdec   = (float*)   carve((size_t)BQ_NPIX * sizeof(float));
    _Float16* qh     = (_Float16*)carve((size_t)BQ_TMAX * BQ_C * sizeof(_Float16));
    int*      tokens = (int*)     carve((size_t)BQ_TMAX * sizeof(int));
    float*    zq     = (float*)   carve((size_t)BQ_TMAX * BQ_C * sizeof(float));
    (void)ws_size; (void)in_sizes; (void)n_in; (void)out_size;

    bq_prep_emb<<<(BQ_V + 255) / 256, 256, 0, stream>>>(emb, embh, esq);
    bq_init_z<<<(BQ_NPIX + 255) / 256, 256, 0, stream>>>(z_enc, zrest, zdec);

    const int ms[6] = {1, 2, 4, 8, 16, 32};
    for (int level = 0; level < 6; ++level) {
        const int pn = ms[level];
        const int T  = BQ_B * pn * pn;

        bq_downsample<<<(T * BQ_C + 255) / 256, 256, 0, stream>>>(zrest, qh, pn);

        // 4 waves/block, 32 queries/wave -> 128 queries per block
        int blocks = (T + 127) / 128;
        bq_argmin<<<blocks, 128, 0, stream>>>(qh, embh, esq, tokens, T);

        bq_gather<<<(T * BQ_C + 255) / 256, 256, 0, stream>>>(emb, tokens, zq, T);

        bq_upsample_update<<<(BQ_NPIX + 255) / 256, 256, 0, stream>>>(
            zq, zdec, zrest, out + (size_t)level * BQ_NPIX, pn);
    }
}
